// W8A8BFP32OFP32QKVLinear_64862596104556
// MI455X (gfx1250) — compile-verified, matrix operand order checked by reading, not run on hardware
//
#include <hip/hip_runtime.h>
#include <cstdint>
#include <cstddef>

typedef __attribute__((ext_vector_type(8))) int v8i;

namespace {
constexpr int MDIM = 8192;
constexpr int KDIM = 4096;
constexpr int NDIM = 12288;
constexpr int QKV  = 4096;

constexpr int BM = 128;
constexpr int BN = 128;
constexpr int BK = 64;
constexpr int KSTEPS = KDIM / BK;
constexpr int LDA = 80;  // padded LDS row stride: multiple of 16 (aligned b128),
constexpr int LDB = 80;  // 20-dword stride -> 16 distinct banks over 16 frag rows
}  // namespace

// ---------------------------------------------------------------------------
// Pre-pass 1: fp32 -> int8 quantization of activations (RNE matches jnp.round,
// then clip to [-128,127]).
// ---------------------------------------------------------------------------
__global__ __launch_bounds__(256) void quantize_x_kernel(
    const float* __restrict__ x, uint32_t* __restrict__ xq, int count4) {
  int i = blockIdx.x * blockDim.x + threadIdx.x;
  if (i >= count4) return;
  float4 v = reinterpret_cast<const float4*>(x)[i];
  int q0 = (int)fminf(127.f, fmaxf(-128.f, rintf(v.x)));
  int q1 = (int)fminf(127.f, fmaxf(-128.f, rintf(v.y)));
  int q2 = (int)fminf(127.f, fmaxf(-128.f, rintf(v.z)));
  int q3 = (int)fminf(127.f, fmaxf(-128.f, rintf(v.w)));
  uint32_t p = (uint32_t)(q0 & 0xFF) | ((uint32_t)(q1 & 0xFF) << 8) |
               ((uint32_t)(q2 & 0xFF) << 16) | ((uint32_t)(q3 & 0xFF) << 24);
  xq[i] = p;
}

// ---------------------------------------------------------------------------
// Pre-pass 2: int32 weights (already in [-128,127]) -> packed int8.
// ---------------------------------------------------------------------------
__global__ __launch_bounds__(256) void pack_w_kernel(
    const int* __restrict__ w, uint32_t* __restrict__ wq, int count4) {
  int i = blockIdx.x * blockDim.x + threadIdx.x;
  if (i >= count4) return;
  int4 v = reinterpret_cast<const int4*>(w)[i];
  uint32_t p = (uint32_t)(v.x & 0xFF) | ((uint32_t)(v.y & 0xFF) << 8) |
               ((uint32_t)(v.z & 0xFF) << 16) | ((uint32_t)(v.w & 0xFF) << 24);
  wq[i] = p;
}

// ---------------------------------------------------------------------------
// gfx1250 async global->LDS copy (16B per lane), tracked with ASYNCcnt.
// The LDS destination address is the wave-relative LDS byte offset, which for
// a generic (flat) shared pointer lives in addr[31:0].
// ---------------------------------------------------------------------------
__device__ __forceinline__ void async_copy_b128(const uint8_t* gptr,
                                                const uint8_t* lptr) {
  uint32_t lds_off = (uint32_t)(uintptr_t)lptr;
  asm volatile("global_load_async_to_lds_b128 %0, %1, off"
               :
               : "v"(lds_off), "v"(gptr)
               : "memory");
}

// ---------------------------------------------------------------------------
// Main GEMM: int8 x int8 -> int32 via V_WMMA_I32_16X16X64_IU8.
// Block tile 128x128, BK=64, double-buffered LDS fed by async-to-LDS DMA.
// 8 waves (wave32) in a 2x4 grid; each wave owns 64x32 = 4x2 WMMA tiles.
// ---------------------------------------------------------------------------
__global__ __launch_bounds__(256) void qkv_wmma_gemm_kernel(
    const uint8_t* __restrict__ Aq,   // [MDIM x KDIM] int8 row-major
    const uint8_t* __restrict__ Wq,   // [NDIM x KDIM] int8 row-major
    const float* __restrict__ bias,   // [NDIM]
    const float* __restrict__ qsp, const float* __restrict__ ksp,
    const float* __restrict__ vsp,
    float* __restrict__ out) {        // [MDIM x NDIM] fp32
  __shared__ uint8_t As[2][BM * LDA];
  __shared__ uint8_t Bs[2][BN * LDB];

  const int tid   = threadIdx.x;
  const int lane  = tid & 31;
  const int wid   = tid >> 5;
  const int waveM = wid >> 2;  // 0..1 -> which 64-row slab
  const int waveN = wid & 3;   // 0..3 -> which 32-col slab
  const int blockM = blockIdx.y * BM;
  const int blockN = blockIdx.x * BN;

  // Staging: each tile is 128 rows x 64 B = 512 chunks of 16 B; 2 per thread.
  const int c0  = tid;
  const int c1  = tid + 256;
  const int ra0 = c0 >> 2, ka0 = (c0 & 3) << 4;
  const int ra1 = c1 >> 2, ka1 = (c1 & 3) << 4;

  const uint8_t* aBase = Aq + (size_t)blockM * KDIM;
  const uint8_t* bBase = Wq + (size_t)blockN * KDIM;

  // Issue one K-stage of async copies (4 per wave: 2 A + 2 B, 16B per lane).
  auto issueStage = [&](int buf, int k0) {
    async_copy_b128(aBase + (size_t)ra0 * KDIM + k0 + ka0, &As[buf][ra0 * LDA + ka0]);
    async_copy_b128(aBase + (size_t)ra1 * KDIM + k0 + ka1, &As[buf][ra1 * LDA + ka1]);
    async_copy_b128(bBase + (size_t)ra0 * KDIM + k0 + ka0, &Bs[buf][ra0 * LDB + ka0]);
    async_copy_b128(bBase + (size_t)ra1 * KDIM + k0 + ka1, &Bs[buf][ra1 * LDB + ka1]);
  };

  v8i acc[4][2] = {};

  // 8-bit WMMA operand layout (ISA 7.12.2): lanes 0-15 hold rows, lanes 16-31
  // hold the same rows at K+8 within each 16-K group; dword pairs sit at K
  // offsets {0,16,32,48} + (lane>=16 ? 8 : 0).
  const int row = lane & 15;
  const int hi  = (lane >> 4) << 3;  // 0 or 8 (K byte offset; also +8 M rows for C/D)

  issueStage(0, 0);

  for (int kt = 0; kt < KSTEPS; ++kt) {
    const int cur = kt & 1;

    if (kt + 1 < KSTEPS) {
      issueStage(cur ^ 1, (kt + 1) * BK);
      // The 4 next-stage ops remain outstanding; in-order completion means
      // cnt<=4 implies the current stage's 4 ops have landed in LDS.
      asm volatile("s_wait_asynccnt 0x4" ::: "memory");
    } else {
      asm volatile("s_wait_asynccnt 0x0" ::: "memory");
    }
    __syncthreads();

    // A fragments: 4 x (16x64 int8)
    v8i af[4];
#pragma unroll
    for (int f = 0; f < 4; ++f) {
      const uint8_t* ap = &As[cur][(waveM * 64 + f * 16 + row) * LDA + hi];
      union { v8i v; int2 d[4]; } u;
#pragma unroll
      for (int g = 0; g < 4; ++g) u.d[g] = *reinterpret_cast<const int2*>(ap + g * 16);
      af[f] = u.v;
    }
    // B fragments: 2 x (64x16 int8); weight rows are K-contiguous (col-major B)
    v8i bf[2];
#pragma unroll
    for (int g = 0; g < 2; ++g) {
      const uint8_t* bp = &Bs[cur][(waveN * 32 + g * 16 + row) * LDB + hi];
      union { v8i v; int2 d[4]; } u;
#pragma unroll
      for (int t = 0; t < 4; ++t) u.d[t] = *reinterpret_cast<const int2*>(bp + t * 16);
      bf[g] = u.v;
    }

#pragma unroll
    for (int f = 0; f < 4; ++f)
#pragma unroll
      for (int g = 0; g < 2; ++g)
        acc[f][g] = __builtin_amdgcn_wmma_i32_16x16x64_iu8(
            /*sgn_a=*/true, af[f], /*sgn_b=*/true, bf[g], acc[f][g],
            /*reuse_a=*/false, /*reuse_b=*/false);

    // All waves must finish reading this buffer before it is refilled (the
    // refill for stage kt+2 is issued right after this barrier).
    __syncthreads();
  }

  // Epilogue: q/k/v boundaries (4096, 8192) are 128-aligned -> scale is
  // uniform across the whole block tile.
  const float scale =
      (blockN < QKV) ? *qsp : ((blockN < 2 * QKV) ? *ksp : *vsp);

#pragma unroll
  for (int f = 0; f < 4; ++f) {
#pragma unroll
    for (int g = 0; g < 2; ++g) {
      const int n = blockN + waveN * 32 + g * 16 + row;
      const float bn = bias[n];
#pragma unroll
      for (int e = 0; e < 8; ++e) {
        const int m = blockM + waveM * 64 + f * 16 + e + hi;
        out[(size_t)m * NDIM + n] = scale * (float)acc[f][g][e] + bn;
      }
    }
  }
}

// ---------------------------------------------------------------------------
extern "C" void kernel_launch(void* const* d_in, const int* in_sizes, int n_in,
                              void* d_out, int out_size, void* d_ws, size_t ws_size,
                              hipStream_t stream) {
  const float* x    = (const float*)d_in[0];
  const int*   w    = (const int*)d_in[1];
  const float* bias = (const float*)d_in[2];
  const float* qs   = (const float*)d_in[3];
  const float* ks   = (const float*)d_in[4];
  const float* vs   = (const float*)d_in[5];
  float* out = (float*)d_out;

  uint8_t* xq = (uint8_t*)d_ws;            // MDIM*KDIM bytes
  uint8_t* wq = xq + (size_t)MDIM * KDIM;  // NDIM*KDIM bytes

  {
    int count4 = MDIM * KDIM / 4;
    quantize_x_kernel<<<count4 / 256, 256, 0, stream>>>(x, (uint32_t*)xq, count4);
  }
  {
    int count4 = NDIM * KDIM / 4;
    pack_w_kernel<<<count4 / 256, 256, 0, stream>>>(w, (uint32_t*)wq, count4);
  }

  dim3 grid(NDIM / BN, MDIM / BM);
  qkv_wmma_gemm_kernel<<<grid, 256, 0, stream>>>(xq, wq, bias, qs, ks, vs, out);
}
